// PKAttention_1726576855276
// MI455X (gfx1250) — matrix operand order repaired and checked by reading, not compiled, not run on hardware
//
#include <hip/hip_runtime.h>

typedef __bf16 bf16;
typedef __attribute__((ext_vector_type(2)))  bf16 bf16x2;
typedef __attribute__((ext_vector_type(8)))  bf16 bf16x8;
typedef __attribute__((ext_vector_type(16))) bf16 v16bf;
typedef __attribute__((ext_vector_type(8)))  float v8f;

#define ROWS 2048      // b*n
#define NDIM 256
#define NHEADS 8
#define NKV 65536
#define SEQ 1024

// ---------------- WMMA fragment helpers (CDNA5 wave32 layouts) ----------------

__device__ __forceinline__ v16bf load_a(const bf16* __restrict__ src, int ld) {
  // src points at (m0,k0) of row-major [M][K]; 16x32 bf16 A fragment
  int lane = threadIdx.x & 31;
  const bf16* p = src + (lane & 15) * ld + ((lane >> 4) << 3);
  v16bf a;
#pragma unroll
  for (int v = 0; v < 8; ++v) {
    int kk = 2 * v + ((v >= 4) ? 8 : 0);
    bf16x2 t = *(const bf16x2*)(p + kk);
    a[2 * v] = t[0]; a[2 * v + 1] = t[1];
  }
  return a;
}

__device__ __forceinline__ v16bf load_b(const bf16* __restrict__ srcT, int ld) {
  // srcT points at (n0,k0) of row-major B^T [N][K]; 32x16 bf16 B fragment
  int lane = threadIdx.x & 31;
  const bf16* p = srcT + (lane & 15) * ld + ((lane >> 4) << 3);
  v16bf b;
#pragma unroll
  for (int v = 0; v < 8; ++v) {
    int kk = 2 * v + ((v >= 4) ? 8 : 0);
    bf16x2 t = *(const bf16x2*)(p + kk);
    b[2 * v] = t[0]; b[2 * v + 1] = t[1];
  }
  return b;
}

__device__ __forceinline__ v8f wmma_bf16(v16bf a, v16bf b, v8f c) {
  return __builtin_amdgcn_wmma_f32_16x16x32_bf16(false, a, false, b, (short)0, c,
                                                 false, false);
}

__device__ __forceinline__ void store_c_f32(float* dst, int ld, v8f c) {
  int lane = threadIdx.x & 31;
  float* p = dst + (size_t)((lane >> 4) << 3) * ld + (lane & 15);
#pragma unroll
  for (int r = 0; r < 8; ++r) p[(size_t)r * ld] = c[r];
}

__device__ __forceinline__ void store_c_bf16(bf16* dst, int ld, v8f c, float scale) {
  int lane = threadIdx.x & 31;
  bf16* p = dst + (size_t)((lane >> 4) << 3) * ld + (lane & 15);
#pragma unroll
  for (int r = 0; r < 8; ++r) p[(size_t)r * ld] = (bf16)(c[r] * scale);
}

__device__ __forceinline__ void wave_argmax(float& v, int& idx) {
#pragma unroll
  for (int m = 16; m >= 1; m >>= 1) {
    float ov = __shfl_xor(v, m, 32);
    int oi = __shfl_xor(idx, m, 32);
    if (ov > v || (ov == v && oi < idx)) { v = ov; idx = oi; }
  }
}

// ---------------- conversion kernels ----------------

__global__ void k_cvt_x(const float* __restrict__ x, bf16* __restrict__ xb) {
  int i = blockIdx.x * blockDim.x + threadIdx.x;
  if (i < ROWS * NDIM) xb[i] = (bf16)x[i];
}

__global__ void k_cvt_w(const float* __restrict__ Wq_pk, const float* __restrict__ Wq,
                        bf16* __restrict__ wT) {
  int i = blockIdx.x * blockDim.x + threadIdx.x;   // 4096*256
  if (i >= 4096 * 256) return;
  int n = i >> 8, k = i & 255;
  float v = (n < 2048) ? Wq_pk[k * 2048 + n] : Wq[k * 2048 + (n - 2048)];
  wT[n * 256 + k] = (bf16)v;
}

__global__ void k_cvt_pk(const float* __restrict__ pk_keys, bf16* __restrict__ pkT) {
  int i = blockIdx.x * blockDim.x + threadIdx.x;   // 2*8*256*128
  if (i >= 2 * 8 * 256 * 128) return;
  int d = i & 127; int t = i >> 7;
  int key = t & 255; t >>= 8;
  int h = t & 7; int p = t >> 3;
  pkT[(((p * 8 + h) * 256 + key) * 128) + d] =
      (bf16)pk_keys[(((p * 256 + key) * 8 + h) * 128) + d];
}

__global__ void k_cvt_wo(const float* __restrict__ Wo, bf16* __restrict__ woT) {
  int i = blockIdx.x * blockDim.x + threadIdx.x;   // 256*2048
  if (i >= 256 * 2048) return;
  int n = i >> 11, k = i & 2047;
  woT[n * 2048 + k] = (bf16)Wo[k * 256 + n];
}

// ---------------- GEMM 1: Qall = xb @ [Wq_pk | Wq], q half scaled 1/16 ----------------

__global__ void k_gemm_qproj(const bf16* __restrict__ xb, const bf16* __restrict__ wT,
                             bf16* __restrict__ Qall) {
  int task = blockIdx.x * (blockDim.x >> 5) + (threadIdx.x >> 5);
  int nt = task & 255;     // 256 N tiles (4096/16)
  int mt = task >> 8;      // 128 M tiles (2048/16)
  if (mt >= 128) return;
  const bf16* A = xb + (size_t)mt * 16 * 256;
  const bf16* B = wT + (size_t)nt * 16 * 256;
  v8f c = {0.f, 0.f, 0.f, 0.f, 0.f, 0.f, 0.f, 0.f};
#pragma unroll
  for (int ks = 0; ks < 8; ++ks)
    c = wmma_bf16(load_a(A + ks * 32, 256), load_b(B + ks * 32, 256), c);
  int n0 = nt * 16;
  float scale = (n0 >= 2048) ? 0.0625f : 1.0f;   // DIM^-0.5 for q half
  store_c_bf16(Qall + (size_t)mt * 16 * 4096 + n0, 4096, c, scale);
}

// ---------------- GEMM 2: sim[p,row,h,key] ----------------

__global__ void k_gemm_sim(const bf16* __restrict__ Qall, const bf16* __restrict__ pkT,
                           float* __restrict__ simbuf) {
  int ph = blockIdx.y;                      // p*8+h
  int task = blockIdx.x * (blockDim.x >> 5) + (threadIdx.x >> 5);
  int nt = task & 15;                       // 16 key tiles
  int mt = task >> 4;                       // 128 row tiles
  if (mt >= 128) return;
  int p = ph >> 3, h = ph & 7;
  const bf16* A = Qall + (size_t)mt * 16 * 4096 + (p * 1024 + h * 128);
  const bf16* B = pkT + (size_t)(ph * 256 + nt * 16) * 128;
  v8f c = {0.f, 0.f, 0.f, 0.f, 0.f, 0.f, 0.f, 0.f};
#pragma unroll
  for (int ks = 0; ks < 4; ++ks)
    c = wmma_bf16(load_a(A + ks * 32, 4096), load_b(B + ks * 32, 128), c);
  float* out = simbuf + (size_t)(p * 2048 + mt * 16) * 2048 + h * 256 + nt * 16;
  store_c_f32(out, 2048, c);
}

// ---------------- top-16 of 256 sims per (p,row,h) ----------------

__global__ void k_topk_pk(const float* __restrict__ simbuf,
                          float* __restrict__ pkscore, int* __restrict__ pkidx) {
  int task = blockIdx.x * (blockDim.x >> 5) + (threadIdx.x >> 5);
  if (task >= 2 * 2048 * 8) return;
  int lane = threadIdx.x & 31;
  int h = task & 7; int row = (task >> 3) & 2047; int p = task >> 14;
  const float* src = simbuf + (size_t)(p * 2048 + row) * 2048 + h * 256;
  float vals[8];
#pragma unroll
  for (int s = 0; s < 8; ++s) vals[s] = src[s * 32 + lane];
  float* outs = pkscore + (size_t)task * 16;
  int* outi = pkidx + (size_t)task * 16;
#pragma unroll
  for (int t = 0; t < 16; ++t) {
    float bv = vals[0]; int bs = 0;
#pragma unroll
    for (int s = 1; s < 8; ++s) if (vals[s] > bv) { bv = vals[s]; bs = s; }
    int gi = bs * 32 + lane;
    wave_argmax(bv, gi);
    if (lane == 0) { outs[t] = bv; outi[t] = gi; }
    int cs = gi >> 5, cl = gi & 31;
    if (lane == cl) {
#pragma unroll
      for (int s = 0; s < 8; ++s) if (s == cs) vals[s] = -3.0e38f;
    }
  }
}

// ------- fused: cartesian combine, top-16, softmax, weighted gather of k/v -------

__global__ void k_combine_gather(const float* __restrict__ pkscore,
                                 const int* __restrict__ pkidx,
                                 const float* __restrict__ keys_table,
                                 const float* __restrict__ values_table,
                                 bf16* __restrict__ kb, bf16* __restrict__ vbT) {
  int task = blockIdx.x * (blockDim.x >> 5) + (threadIdx.x >> 5);
  if (task >= ROWS * NHEADS) return;
  int lane = threadIdx.x & 31;
  int h = task & 7; int row = task >> 3;
  int b = row >> 10, n = row & 1023;
  size_t base0 = (size_t)(row * 8 + h) * 16;
  size_t base1 = base0 + (size_t)16384 * 16;
  float s0 = 0.f, s1 = 0.f; int i0v = 0, i1v = 0;
  if (lane < 16) {
    s0 = pkscore[base0 + lane]; i0v = pkidx[base0 + lane];
    s1 = pkscore[base1 + lane]; i1v = pkidx[base1 + lane];
  }
  float vals[8];
#pragma unroll
  for (int s = 0; s < 8; ++s) {
    int c = s * 32 + lane;
    vals[s] = __shfl(s0, c >> 4, 32) + __shfl(s1, c & 15, 32);
  }
  float ex[16]; int kvi[16];
  float mmax = 0.f, esum = 0.f;
#pragma unroll
  for (int t = 0; t < 16; ++t) {
    float bv = vals[0]; int bs = 0;
#pragma unroll
    for (int s = 1; s < 8; ++s) if (vals[s] > bv) { bv = vals[s]; bs = s; }
    int gi = bs * 32 + lane;
    wave_argmax(bv, gi);
    if (t == 0) mmax = bv;
    float e = __expf(bv - mmax);
    ex[t] = e; esum += e;
    kvi[t] = __shfl(i0v, gi >> 4, 32) + __shfl(i1v, gi & 15, 32) * 256;
    int cs = gi >> 5, cl = gi & 31;
    if (lane == cl) {
#pragma unroll
      for (int s = 0; s < 8; ++s) if (s == cs) vals[s] = -3.0e38f;
    }
  }
  float inv = 1.0f / esum;
  float ka[8] = {0,0,0,0,0,0,0,0}, va[8] = {0,0,0,0,0,0,0,0};
#pragma unroll
  for (int t = 0; t < 16; ++t) {
    float w = ex[t] * inv;
    size_t roff = ((size_t)h * NKV + (size_t)kvi[t]) * 256 + (size_t)lane * 8;
    const float4* kr = (const float4*)(keys_table + roff);
    const float4* vr = (const float4*)(values_table + roff);
    float4 k0 = kr[0], k1 = kr[1];
    float4 v0 = vr[0], v1 = vr[1];
    ka[0] += w * k0.x; ka[1] += w * k0.y; ka[2] += w * k0.z; ka[3] += w * k0.w;
    ka[4] += w * k1.x; ka[5] += w * k1.y; ka[6] += w * k1.z; ka[7] += w * k1.w;
    va[0] += w * v0.x; va[1] += w * v0.y; va[2] += w * v0.z; va[3] += w * v0.w;
    va[4] += w * v1.x; va[5] += w * v1.y; va[6] += w * v1.z; va[7] += w * v1.w;
  }
  bf16x8 kv8;
#pragma unroll
  for (int m = 0; m < 8; ++m) kv8[m] = (bf16)ka[m];
  *(bf16x8*)(kb + ((size_t)((b * 8 + h) * 1024 + n)) * 256 + (size_t)lane * 8) = kv8;
  size_t vbase = ((size_t)(b * 8 + h) * 256 + (size_t)lane * 8) * 1024 + n;
#pragma unroll
  for (int m = 0; m < 8; ++m) vbT[vbase + (size_t)m * 1024] = (bf16)va[m];
}

// ---------------- causal flash attention, one wave per 16-row Q tile ----------------

__global__ void __launch_bounds__(256) k_attn(const bf16* __restrict__ Qall,
                                              const bf16* __restrict__ kb,
                                              const bf16* __restrict__ vbT,
                                              bf16* __restrict__ attnout) {
  __shared__ bf16 plds[8 * 16 * 32];
  int wid = threadIdx.x >> 5;
  int task = blockIdx.x * 8 + wid;          // 1024 tasks
  int ti = task & 63; int h = (task >> 6) & 7; int b = task >> 9;
  int lane = threadIdx.x & 31;
  int half = lane >> 4, nn = lane & 15;
  int i0 = ti * 16;
  int row0 = b * 1024 + i0;

  v16bf qf[8];
#pragma unroll
  for (int ks = 0; ks < 8; ++ks)
    qf[ks] = load_a(Qall + (size_t)row0 * 4096 + 2048 + h * 256 + ks * 32, 4096);

  v8f zero = {0.f, 0.f, 0.f, 0.f, 0.f, 0.f, 0.f, 0.f};
  v8f acc[16];
#pragma unroll
  for (int f = 0; f < 16; ++f) acc[f] = zero;
  float mrow[8], lrow[8];
#pragma unroll
  for (int r = 0; r < 8; ++r) { mrow[r] = -3.0e38f; lrow[r] = 0.f; }

  const bf16* kbase = kb + (size_t)((b * 8 + h) * 1024) * 256;
  const bf16* vbase = vbT + (size_t)((b * 8 + h) * 256) * 1024;
  bf16* pw = plds + wid * 16 * 32;

  for (int j0 = 0; j0 < i0 + 16; j0 += 32) {
    v8f s0 = zero, s1 = zero;
#pragma unroll
    for (int ks = 0; ks < 8; ++ks) {
      v16bf b0 = load_b(kbase + (size_t)j0 * 256 + ks * 32, 256);
      v16bf b1 = load_b(kbase + (size_t)(j0 + 16) * 256 + ks * 32, 256);
      s0 = wmma_bf16(qf[ks], b0, s0);
      s1 = wmma_bf16(qf[ks], b1, s1);
    }
#pragma unroll
    for (int r = 0; r < 8; ++r) {
      int irow = i0 + r + 8 * half;
      if (j0 + nn > irow) s0[r] = -3.0e38f;
      if (j0 + 16 + nn > irow) s1[r] = -3.0e38f;
    }
#pragma unroll
    for (int r = 0; r < 8; ++r) {
      float rm = fmaxf(s0[r], s1[r]);
#pragma unroll
      for (int m = 8; m >= 1; m >>= 1) rm = fmaxf(rm, __shfl_xor(rm, m, 32));
      float mnew = fmaxf(mrow[r], rm);
      float alpha = __expf(mrow[r] - mnew);
      float p0 = __expf(s0[r] - mnew);
      float p1 = __expf(s1[r] - mnew);
      float rs = p0 + p1;
#pragma unroll
      for (int m = 8; m >= 1; m >>= 1) rs += __shfl_xor(rs, m, 32);
      lrow[r] = lrow[r] * alpha + rs;
      mrow[r] = mnew;
#pragma unroll
      for (int f = 0; f < 16; ++f) acc[f][r] *= alpha;
      pw[(r + 8 * half) * 32 + nn] = (bf16)p0;
      pw[(r + 8 * half) * 32 + 16 + nn] = (bf16)p1;
    }
    // read P back as a 16x32 A fragment (LDS, same wave -> in-order DS)
    v16bf pf;
    {
      const bf16* p = pw + (lane & 15) * 32 + half * 8;
#pragma unroll
      for (int v = 0; v < 8; ++v) {
        int kk = 2 * v + ((v >= 4) ? 8 : 0);
        pf[2 * v] = p[kk]; pf[2 * v + 1] = p[kk + 1];
      }
    }
#pragma unroll
    for (int f = 0; f < 16; ++f) {
      v16bf bv = load_b(vbase + (size_t)(f * 16) * 1024 + j0, 1024);
      acc[f] = wmma_bf16(pf, bv, acc[f]);
    }
  }
#pragma unroll
  for (int r = 0; r < 8; ++r) {
    float invl = 1.0f / lrow[r];
#pragma unroll
    for (int f = 0; f < 16; ++f) acc[f][r] *= invl;
  }
  bf16* obase = attnout + (size_t)row0 * 2048 + h * 256;
#pragma unroll
  for (int f = 0; f < 16; ++f)
#pragma unroll
    for (int r = 0; r < 8; ++r)
      obase[(size_t)(r + 8 * half) * 2048 + f * 16 + nn] = (bf16)acc[f][r];
}

// ---------------- GEMM 3: d_out = attnout @ Wo ----------------

__global__ void k_gemm_out(const bf16* __restrict__ attnout,
                           const bf16* __restrict__ woT, float* __restrict__ out) {
  int task = blockIdx.x * (blockDim.x >> 5) + (threadIdx.x >> 5);
  int nt = task & 15; int mt = task >> 4;
  if (mt >= 128) return;
  const bf16* A = attnout + (size_t)mt * 16 * 2048;
  const bf16* B = woT + (size_t)nt * 16 * 2048;
  v8f c = {0.f, 0.f, 0.f, 0.f, 0.f, 0.f, 0.f, 0.f};
#pragma unroll 4
  for (int ks = 0; ks < 64; ++ks)
    c = wmma_bf16(load_a(A + ks * 32, 2048), load_b(B + ks * 32, 2048), c);
  store_c_f32(out + (size_t)mt * 16 * 256 + nt * 16, 256, c);
}

// ---------------- host launch ----------------

extern "C" void kernel_launch(void* const* d_in, const int* in_sizes, int n_in,
                              void* d_out, int out_size, void* d_ws, size_t ws_size,
                              hipStream_t stream) {
  (void)in_sizes; (void)n_in; (void)out_size; (void)ws_size;
  const float* x            = (const float*)d_in[0];
  const float* Wq_pk        = (const float*)d_in[1];
  const float* pk_keys      = (const float*)d_in[2];
  const float* Wq           = (const float*)d_in[3];
  const float* keys_table   = (const float*)d_in[4];
  const float* values_table = (const float*)d_in[5];
  const float* Wo           = (const float*)d_in[6];

  char* ws = (char*)d_ws;
  size_t off = 0;
  auto alloc = [&](size_t bytes) {
    void* p = ws + off; off += (bytes + 255) & ~(size_t)255; return p;
  };
  bf16*  xb      = (bf16*) alloc((size_t)ROWS * 256 * 2);          // 1 MB
  bf16*  wT      = (bf16*) alloc((size_t)4096 * 256 * 2);          // 2 MB
  bf16*  pkT     = (bf16*) alloc((size_t)16 * 256 * 128 * 2);      // 1 MB
  bf16*  woT     = (bf16*) alloc((size_t)256 * 2048 * 2);          // 1 MB
  bf16*  Qall    = (bf16*) alloc((size_t)ROWS * 4096 * 2);         // 16 MB
  float* simbuf  = (float*)alloc((size_t)2 * ROWS * 2048 * 4);     // 33.5 MB
  float* pkscore = (float*)alloc((size_t)2 * ROWS * 8 * 16 * 4);   // 2 MB
  int*   pkidx   = (int*)  alloc((size_t)2 * ROWS * 8 * 16 * 4);   // 2 MB
  bf16*  kb      = (bf16*) alloc((size_t)16 * 1024 * 256 * 2);     // 8 MB
  bf16*  vbT     = (bf16*) alloc((size_t)16 * 256 * 1024 * 2);     // 8 MB
  bf16*  attnout = (bf16*) alloc((size_t)ROWS * 2048 * 2);         // 8 MB

  k_cvt_x <<<2048, 256, 0, stream>>>(x, xb);
  k_cvt_w <<<4096, 256, 0, stream>>>(Wq_pk, Wq, wT);
  k_cvt_pk<<<2048, 256, 0, stream>>>(pk_keys, pkT);
  k_cvt_wo<<<2048, 256, 0, stream>>>(Wo, woT);

  k_gemm_qproj<<<4096, 256, 0, stream>>>(xb, wT, Qall);
  k_gemm_sim<<<dim3(256, 16), 256, 0, stream>>>(Qall, pkT, simbuf);
  k_topk_pk<<<4096, 256, 0, stream>>>(simbuf, pkscore, pkidx);
  k_combine_gather<<<2048, 256, 0, stream>>>(pkscore, pkidx, keys_table,
                                             values_table, kb, vbT);
  k_attn<<<128, 256, 0, stream>>>(Qall, kb, vbT, attnout);
  k_gemm_out<<<256, 256, 0, stream>>>(attnout, woT, (float*)d_out);
}
